// MultiHeadAttention_51651276701936
// MI455X (gfx1250) — compile-verified, matrix operand order checked by reading, not run on hardware
//
#include <hip/hip_runtime.h>
#include <hip/hip_bf16.h>

// MI455X / gfx1250 fused multi-head attention:
//   k1: QKV GEMM  (x f32 @ w_qkv^T + b_qkv) -> qkv bf16   [8192 x 3072]
//   k2: flash attention per (b,h)           -> attn bf16  [8192 x 1024]
//   k3: out GEMM  (attn bf16 @ w_out^T + b) -> out f32    [8192 x 1024]
// Matrix math via v_wmma_f32_16x16x32_bf16 (wave32); K-block staging in the
// attention kernel uses the Tensor Data Mover (tensor_load_to_lds + TENSORcnt).

typedef __attribute__((ext_vector_type(16))) __bf16 v16bf;
typedef __attribute__((ext_vector_type(8)))  float  v8f;
typedef __attribute__((ext_vector_type(4)))  unsigned v4u;
typedef __attribute__((ext_vector_type(8)))  int      v8i;
typedef __attribute__((ext_vector_type(4)))  int      v4i;

#define D_MODEL   1024
#define QKV_DIM   3072
#define SEQ       2048
#define BATCH     4
#define HEADS     16
#define HEAD_DIM  64

// 0.125 (1/sqrt(64)) * log2(e): folds attention scale + base-2 exp into one FMA
#define SCALE_LOG2E 0.1803368801f

// native f32 -> bf16 (backend selects v_cvt_*bf16_f32 on gfx1250)
__device__ __forceinline__ __bf16 f2bf(float f) {
    __hip_bfloat16 h = __float2bfloat16(f);
    return __builtin_bit_cast(__bf16, h);
}

// WMMA D = A*B + C (bf16 inputs, f32 accumulate)
__device__ __forceinline__ v8f wmma_bf16(v16bf a, v16bf b, v8f c) {
    return __builtin_amdgcn_wmma_f32_16x16x32_bf16(false, a, false, b,
                                                   (short)0, c, false, false);
}

// Load a 16x32 bf16 fragment (A: lane=row / B: lane=col), row-major storage,
// stride `ld` elements. Element layout per CDNA5 ISA 7.12.2:
//   element e = 2v+lo holds K = 16*(v>>2) + 2*(v&3) + 8*(lane>=16) + lo
__device__ __forceinline__ v16bf frag_ld(const __bf16* base, int ld) {
    const int lane = threadIdx.x & 31;
    const int row  = lane & 15;
    const int g    = lane >> 4;
    v16bf f;
#pragma unroll
    for (int v = 0; v < 8; ++v) {
        const int k = ((v >> 2) << 4) + ((v & 3) << 1) + (g << 3);
        f[2 * v]     = base[row * ld + k];
        f[2 * v + 1] = base[row * ld + k + 1];
    }
    return f;
}

// ---------------------------------------------------------------------------
// GEMM: C[M,N] = A[M,K] @ W[N,K]^T + bias[N]
// Block = 256 threads (8 waves), tile 128x128, wave tile 32x64 (2x4 WMMA).
// Software-pipelined: next K-tile global loads overlap current WMMAs.
// ---------------------------------------------------------------------------
template <bool A_F32, bool OUT_BF16>
__global__ __launch_bounds__(256, 2)
void gemm_bias_kernel(const void* __restrict__ Av,
                      const float* __restrict__ W,
                      const float* __restrict__ bias,
                      void* __restrict__ outv,
                      int M, int N, int K) {
    __shared__ __align__(16) __bf16 As[128 * 40];
    __shared__ __align__(16) __bf16 Bs[128 * 40];

    const int tid  = threadIdx.x;
    const int bm   = blockIdx.y * 128;
    const int bn   = blockIdx.x * 128;
    const int wid  = tid >> 5;
    const int lane = tid & 31;
    const int wm   = wid & 3;      // 4 waves along M (32 rows each)
    const int wn   = wid >> 2;     // 2 waves along N (64 cols each)
    const int g    = lane >> 4;
    const int rl   = lane & 15;

    v8f acc[2][4] = {};

    float          arf[16];        // staged A (f32 path)
    unsigned short arh[16];        // staged A (bf16 path, raw bits)
    float          brf[16];        // staged W

    // prologue: stage K-tile 0 into registers
#pragma unroll
    for (int i = 0; i < 16; ++i) {
        const int idx = tid + i * 256;                // 0..4095 -> 128x32
        const int r = idx >> 5, c = idx & 31;
        if constexpr (A_F32)
            arf[i] = ((const float*)Av)[(size_t)(bm + r) * K + c];
        else
            arh[i] = ((const unsigned short*)Av)[(size_t)(bm + r) * K + c];
        brf[i] = W[(size_t)(bn + r) * K + c];
    }

    for (int k0 = 0; k0 < K; k0 += 32) {
        // commit staged registers to LDS
#pragma unroll
        for (int i = 0; i < 16; ++i) {
            const int idx = tid + i * 256;
            const int r = idx >> 5, c = idx & 31;
            if constexpr (A_F32)
                As[r * 40 + c] = f2bf(arf[i]);
            else
                As[r * 40 + c] = __builtin_bit_cast(__bf16, arh[i]);
            Bs[r * 40 + c] = f2bf(brf[i]);
        }
        __syncthreads();

        // issue next K-tile global loads (overlap with WMMA below)
        if (k0 + 32 < K) {
#pragma unroll
            for (int i = 0; i < 16; ++i) {
                const int idx = tid + i * 256;
                const int r = idx >> 5, c = idx & 31;
                if constexpr (A_F32)
                    arf[i] = ((const float*)Av)[(size_t)(bm + r) * K + k0 + 32 + c];
                else
                    arh[i] = ((const unsigned short*)Av)[(size_t)(bm + r) * K + k0 + 32 + c];
                brf[i] = W[(size_t)(bn + r) * K + k0 + 32 + c];
            }
        }

        const v16bf a0 = frag_ld(&As[(wm * 32 +  0) * 40], 40);
        const v16bf a1 = frag_ld(&As[(wm * 32 + 16) * 40], 40);
#pragma unroll
        for (int j = 0; j < 4; ++j) {
            const v16bf bj = frag_ld(&Bs[(wn * 64 + j * 16) * 40], 40);
            acc[0][j] = wmma_bf16(a0, bj, acc[0][j]);
            acc[1][j] = wmma_bf16(a1, bj, acc[1][j]);
        }
        __syncthreads();
    }

    // epilogue: add bias, store (C layout: lane=col, elem r -> row r+8g)
#pragma unroll
    for (int tj = 0; tj < 4; ++tj) {
        const int col = bn + wn * 64 + tj * 16 + rl;
        const float bv = bias[col];
#pragma unroll
        for (int ti = 0; ti < 2; ++ti) {
#pragma unroll
            for (int r = 0; r < 8; ++r) {
                const int row = bm + wm * 32 + ti * 16 + r + 8 * g;
                const float v = acc[ti][tj][r] + bv;
                if constexpr (OUT_BF16)
                    ((__bf16*)outv)[(size_t)row * N + col] = f2bf(v);
                else
                    ((float*)outv)[(size_t)row * N + col] = v;
            }
        }
    }
}

// ---------------------------------------------------------------------------
// Flash attention. qkv layout per row: head h occupies cols [h*192, h*192+192)
// with Q = +0..63, K = +64..127, V = +128..191 (matches reference reshape/split).
// Block: (b, h, 128-query tile), 8 waves x 16 queries. Keys in blocks of 128.
// K-block staged by the Tensor Data Mover (one DMA descriptor per block,
// issued by wave 0; TDM pad feature produces the 72-elem padded LDS rows).
// ---------------------------------------------------------------------------
__global__ __launch_bounds__(256, 2)
void flash_attn_kernel(const void* __restrict__ qkv_v,
                       void* __restrict__ out_v) {
    const __bf16* qkv = (const __bf16*)qkv_v;
    __bf16* out = (__bf16*)out_v;

    __shared__ __align__(16) __bf16 Ks[128 * 72];     // [key][d]       18.4 KB
    __shared__ __align__(16) __bf16 Vt[64 * 136];     // [d][key]       17.4 KB
    __shared__ __align__(16) __bf16 Pb[8][16 * 136];  // per-wave P     34.8 KB

    const int b   = blockIdx.z;
    const int h   = blockIdx.y;
    const int q0  = blockIdx.x * 128;
    const int tid = threadIdx.x;
    const int wid = tid >> 5;
    const int lane = tid & 31;
    const int g   = lane >> 4;
    const int rl  = lane & 15;

    const size_t rowbase = (size_t)(b * SEQ) * QKV_DIM + h * 192;
    const __bf16* qg = qkv + rowbase + (size_t)(q0 + wid * 16) * QKV_DIM;
    const __bf16* kg = qkv + rowbase + 64;
    const __bf16* vg = qkv + rowbase + 128;

    // LDS byte offset of Ks for the TDM descriptor (low 32 bits of flat addr)
    const unsigned ldsK = (unsigned)(unsigned long long)&Ks[0];

    // Q fragments live in registers for the whole kernel (stride = 3072)
    const v16bf qf0 = frag_ld(qg, QKV_DIM);
    const v16bf qf1 = frag_ld(qg + 32, QKV_DIM);

    float m[8], l[8];
#pragma unroll
    for (int r = 0; r < 8; ++r) { m[r] = -3.0e38f; l[r] = 0.0f; }
    v8f ao[4] = {};

    for (int j0 = 0; j0 < SEQ; j0 += 128) {
        // --- TDM: DMA K block (128 keys x 64 bf16, row stride 3072 elems)
        // into LDS with padding: 32 DWORDs data + 4 DWORDs pad = 72-elem rows.
        if (wid == 0) {
            const unsigned long long ga =
                (unsigned long long)(kg + (size_t)j0 * QKV_DIM);
            // D# group0: count=1 | lds_addr | global_addr[56:0] | type=2
            const v4u g0 = { 1u,
                             ldsK,
                             (unsigned)(ga & 0xffffffffull),
                             (unsigned)((ga >> 32) & 0x01ffffffull) | 0x80000000u };
            // D# group1: mask=0, data_size=2B, pad_enable, pad_interval=32dw,
            // pad_amount=4dw, tensor_dim0=64, tensor_dim1=2048,
            // tile_dim0=64, tile_dim1=128, tensor_dim0_stride=3072
            const v8i g1 = { (1 << 16) | (1 << 20) | (4 << 22) | (3 << 25),
                             (64 << 16),          // tensor_dim0[15:0]=64
                             (SEQ << 16),         // tensor_dim1[15:0]=2048
                             (64 << 16),          // tile_dim0=64
                             128,                 // tile_dim1=128, tile_dim2=0
                             QKV_DIM,             // dim0_stride low32 = 3072
                             0, 0 };              // dim0_stride hi, dim1_stride
            const v4i gz4 = { 0, 0, 0, 0 };
            const v8i gz8 = { 0, 0, 0, 0, 0, 0, 0, 0 };
            __builtin_amdgcn_tensor_load_to_lds(g0, g1, gz4, gz4, gz8, 0);
            __builtin_amdgcn_s_wait_tensorcnt(0);
        }
        // stage V block transposed: Vt[d][key]
#pragma unroll
        for (int i = 0; i < 16; ++i) {
            const int idx = tid + i * 256;            // 0..4095
            const int kr = idx >> 5, dp = idx & 31;
            const unsigned w =
                *(const unsigned*)(vg + (size_t)(j0 + kr) * QKV_DIM + dp * 2);
            Vt[(dp * 2    ) * 136 + kr] = __builtin_bit_cast(__bf16, (unsigned short)(w & 0xffffu));
            Vt[(dp * 2 + 1) * 136 + kr] = __builtin_bit_cast(__bf16, (unsigned short)(w >> 16));
        }
        if (j0 + 128 < SEQ)
            __builtin_prefetch((const char*)(vg + (size_t)(j0 + 128) * QKV_DIM), 0, 1);
        __syncthreads();

        // raw scores S = Q K^T (16 queries x 128 keys per wave); scale folded
        // into the exponent below.
        v8f sc[8];
#pragma unroll
        for (int t = 0; t < 8; ++t) {
            const v16bf kf0 = frag_ld(&Ks[(t * 16) * 72], 72);
            const v16bf kf1 = frag_ld(&Ks[(t * 16) * 72 + 32], 72);
            v8f s = {};
            s = wmma_bf16(qf0, kf0, s);
            s = wmma_bf16(qf1, kf1, s);
            sc[t] = s;
        }

        // online softmax on raw scores: row max across 16 lanes of each half
        float mn[8];
#pragma unroll
        for (int r = 0; r < 8; ++r) mn[r] = m[r];
#pragma unroll
        for (int t = 0; t < 8; ++t)
#pragma unroll
            for (int r = 0; r < 8; ++r)
                mn[r] = fmaxf(mn[r], sc[t][r]);
#pragma unroll
        for (int off = 1; off < 16; off <<= 1)
#pragma unroll
            for (int r = 0; r < 8; ++r)
                mn[r] = fmaxf(mn[r], __shfl_xor(mn[r], off, 32));

        float corr[8], rs[8];
#pragma unroll
        for (int r = 0; r < 8; ++r) {
            corr[r] = exp2f((m[r] - mn[r]) * SCALE_LOG2E);
            m[r] = mn[r];
            rs[r] = 0.0f;
        }

        // p = 2^((s-mn)*scale*log2e); spill P to per-wave LDS (C->A layout)
        __bf16* pb = Pb[wid];
#pragma unroll
        for (int t = 0; t < 8; ++t)
#pragma unroll
            for (int r = 0; r < 8; ++r) {
                const float p = exp2f((sc[t][r] - mn[r]) * SCALE_LOG2E);
                rs[r] += p;
                pb[(r + 8 * g) * 136 + t * 16 + rl] = f2bf(p);
            }
#pragma unroll
        for (int off = 1; off < 16; off <<= 1)
#pragma unroll
            for (int r = 0; r < 8; ++r)
                rs[r] += __shfl_xor(rs[r], off, 32);
#pragma unroll
        for (int r = 0; r < 8; ++r) l[r] = l[r] * corr[r] + rs[r];
#pragma unroll
        for (int t = 0; t < 4; ++t)
#pragma unroll
            for (int r = 0; r < 8; ++r) ao[t][r] *= corr[r];

        // O += P V  (P as A-fragment from scratch; V via transposed stage)
        v16bf pf[4];
#pragma unroll
        for (int kh = 0; kh < 4; ++kh)
            pf[kh] = frag_ld(pb + kh * 32, 136);
#pragma unroll
        for (int t = 0; t < 4; ++t) {
#pragma unroll
            for (int kh = 0; kh < 4; ++kh) {
                const v16bf vf = frag_ld(&Vt[(t * 16) * 136 + kh * 32], 136);
                ao[t] = wmma_bf16(pf[kh], vf, ao[t]);
            }
        }
        __syncthreads();
    }

    // normalize and store attn_out[b, q, h*64 + d] as bf16
    float inv[8];
#pragma unroll
    for (int r = 0; r < 8; ++r) inv[r] = (l[r] > 0.0f) ? 1.0f / l[r] : 0.0f;
#pragma unroll
    for (int t = 0; t < 4; ++t)
#pragma unroll
        for (int r = 0; r < 8; ++r) {
            const int q = q0 + wid * 16 + r + 8 * g;
            const int d = h * HEAD_DIM + t * 16 + rl;
            out[(size_t)(b * SEQ + q) * D_MODEL + d] = f2bf(ao[t][r] * inv[r]);
        }
}

extern "C" void kernel_launch(void* const* d_in, const int* in_sizes, int n_in,
                              void* d_out, int out_size, void* d_ws, size_t ws_size,
                              hipStream_t stream) {
    (void)in_sizes; (void)n_in; (void)out_size; (void)ws_size;
    const float* x     = (const float*)d_in[0];   // [4,2048,1024]
    const float* w_qkv = (const float*)d_in[1];   // [3072,1024]
    const float* b_qkv = (const float*)d_in[2];   // [3072]
    const float* w_out = (const float*)d_in[3];   // [1024,1024]
    const float* b_out = (const float*)d_in[4];   // [1024]
    float* out = (float*)d_out;                   // [4,2048,1024]

    const int M = BATCH * SEQ;                    // 8192
    // workspace: qkv bf16 (48 MB) then attn_out bf16 (16 MB)
    void* qkv  = d_ws;
    void* attn = (void*)((char*)d_ws + (size_t)M * QKV_DIM * 2);

    dim3 blk(256);
    // QKV projection: [8192,1024] x [3072,1024]^T -> bf16 [8192,3072]
    gemm_bias_kernel<true, true><<<dim3(QKV_DIM / 128, M / 128), blk, 0, stream>>>(
        x, w_qkv, b_qkv, qkv, M, QKV_DIM, D_MODEL);
    // flash attention -> bf16 [8192,1024]
    flash_attn_kernel<<<dim3(SEQ / 128, HEADS, BATCH), blk, 0, stream>>>(qkv, attn);
    // output projection: [8192,1024] x [1024,1024]^T -> f32 out
    gemm_bias_kernel<false, false><<<dim3(D_MODEL / 128, M / 128), blk, 0, stream>>>(
        attn, w_out, b_out, out, M, D_MODEL, D_MODEL);
}
